// ExpertsMLPBlock_39530878992790
// MI455X (gfx1250) — matrix operand order repaired and checked by reading, not compiled, not run on hardware
//
#include <hip/hip_runtime.h>
#include <hip/hip_bf16.h>
#include <math.h>

// Problem constants (from reference setup_inputs)
#define BATCH   8
#define SEQ     4096
#define NTOK    (BATCH * SEQ)      // 32768 tokens
#define CDIM    256                // model dim
#define WH      1024               // hidden dim
#define NEXP    8                  // experts
#define KSEL    2                  // experts per token
#define NPAIR   (NTOK * KSEL)      // 65536 (token, k) pairs

#define TILE_M  64                 // tokens per workgroup
#define WHC     128                // Wh chunk processed per iteration
#define APAD    8                  // LDS padding (halves)

typedef __attribute__((ext_vector_type(16))) __bf16 v16bf;
typedef __attribute__((ext_vector_type(8)))  __bf16 v8bf;
typedef __attribute__((ext_vector_type(8)))  float  v8f;
typedef __attribute__((ext_vector_type(4)))  float  f4;

static constexpr float RMS_EPS = 1.1920928955078125e-07f;

// ---------------- helpers ----------------

__device__ __forceinline__ unsigned short f2bf_bits(float f) {
    union { float f; unsigned int u; } v; v.f = f;
    unsigned int u = v.u;
    unsigned int r = u + 0x7FFFu + ((u >> 16) & 1u);      // round to nearest even
    if ((u & 0x7F800000u) == 0x7F800000u) r = u;          // inf/nan passthrough
    return (unsigned short)(r >> 16);
}

__device__ __forceinline__ __bf16 f2bf(float f) {
    return __builtin_bit_cast(__bf16, f2bf_bits(f));
}

// A fragment (16x32 bf16, MxK) from row-major [.. x ldm] bf16 buffer.
// Lanes 0-15: M=lane, K = k+{0..15}
// Lanes 16-31: M=lane-16, K = k+{16..31}
__device__ __forceinline__ v16bf ldFragA(const __bf16* base, int ldm, int m, int k, int lane) {
    int mm = m + (lane & 15);
    int kk = k + ((lane >> 4) << 4);
    const __bf16* row = base + (size_t)mm * ldm;
    v8bf lo = *(const v8bf*)(row + kk);
    v8bf hi = *(const v8bf*)(row + kk + 8);
    return __builtin_shufflevector(lo, hi, 0, 1, 2, 3, 4, 5, 6, 7,
                                           8, 9, 10, 11, 12, 13, 14, 15);
}

// B fragment (32x16 bf16, KxN) from K-contiguous transposed weights: row n holds
// all K for output column n.  Lanes 0-15: N=lane, K=k+0..15; lanes 16-31: N=lane-16, K=k+16..31.
__device__ __forceinline__ v16bf ldFragB(const __bf16* base, int ldn, int n, int k, int lane) {
    int nn = n + (lane & 15);
    int kk = k + ((lane >> 4) << 4);
    const v8bf* p = (const v8bf*)(base + (size_t)nn * ldn + kk);
    v8bf lo = p[0];
    v8bf hi = p[1];
    return __builtin_shufflevector(lo, hi, 0, 1, 2, 3, 4, 5, 6, 7,
                                           8, 9, 10, 11, 12, 13, 14, 15);
}

// ---------------- kernels ----------------

__global__ void zeroCountsK(int* counts) {
    if (threadIdx.x < NEXP) counts[threadIdx.x] = 0;
}

// One wave per token: sumsq reduce across 32 lanes (8 f32 each), write bf16 xn.
// x is streamed exactly once -> non-temporal loads keep L2 free for weights.
__global__ __launch_bounds__(256) void rmsNormK(const float* __restrict__ x,
                                                const float* __restrict__ w,
                                                __bf16* __restrict__ xn) {
    const int lane = threadIdx.x & 31;
    const int wid  = threadIdx.x >> 5;
    const int tok  = blockIdx.x * 8 + wid;
    if (tok >= NTOK) return;

    const f4* xp = (const f4*)(x + (size_t)tok * CDIM);
    f4 v0 = __builtin_nontemporal_load(xp + lane * 2 + 0);
    f4 v1 = __builtin_nontemporal_load(xp + lane * 2 + 1);
    float ss = v0.x * v0.x + v0.y * v0.y + v0.z * v0.z + v0.w * v0.w +
               v1.x * v1.x + v1.y * v1.y + v1.z * v1.z + v1.w * v1.w;
#pragma unroll
    for (int off = 16; off > 0; off >>= 1) ss += __shfl_xor(ss, off, 32);
    const float scale = rsqrtf(ss * (1.0f / (float)CDIM) + RMS_EPS);

    const f4* wp = (const f4*)w;
    f4 w0 = wp[lane * 2 + 0];
    f4 w1 = wp[lane * 2 + 1];

    v8bf o;
    o[0] = f2bf(v0.x * scale * w0.x); o[1] = f2bf(v0.y * scale * w0.y);
    o[2] = f2bf(v0.z * scale * w0.z); o[3] = f2bf(v0.w * scale * w0.w);
    o[4] = f2bf(v1.x * scale * w1.x); o[5] = f2bf(v1.y * scale * w1.y);
    o[6] = f2bf(v1.z * scale * w1.z); o[7] = f2bf(v1.w * scale * w1.w);
    *(v8bf*)(xn + (size_t)tok * CDIM + lane * 8) = o;
}

// Bucket (token, k) pairs by expert.  Stored value is the output row t*K + k.
__global__ void routeK(const int* __restrict__ ids, int* __restrict__ counts,
                       int* __restrict__ lists) {
    const int i = blockIdx.x * blockDim.x + threadIdx.x;
    if (i >= NPAIR) return;
    const int e = ids[i];
    const int pos = atomicAdd(&counts[e], 1);
    lists[(size_t)e * NPAIR + pos] = i;
}

// W1 [E][C][Wh] f32 -> w1t [E][Wh][C] bf16 (K contiguous per output column)
__global__ void cvtW1K(const float* __restrict__ W1, __bf16* __restrict__ w1t) {
    const size_t i = (size_t)blockIdx.x * blockDim.x + threadIdx.x;
    if (i >= (size_t)NEXP * WH * CDIM) return;
    const int c  = (int)(i % CDIM);
    const int wh = (int)((i / CDIM) % WH);
    const int e  = (int)(i / ((size_t)CDIM * WH));
    w1t[i] = f2bf(W1[((size_t)e * CDIM + c) * WH + wh]);
}

// W2 [E][Wh][C] f32 -> w2t [E][C][Wh] bf16
__global__ void cvtW2K(const float* __restrict__ W2, __bf16* __restrict__ w2t) {
    const size_t i = (size_t)blockIdx.x * blockDim.x + threadIdx.x;
    if (i >= (size_t)NEXP * CDIM * WH) return;
    const int wh = (int)(i % WH);
    const int c  = (int)((i / WH) % CDIM);
    const int e  = (int)(i / ((size_t)WH * CDIM));
    w2t[i] = f2bf(W2[((size_t)e * WH + wh) * CDIM + c]);
}

// Fused per-expert MLP: 64 gathered tokens per block, 8 waves.
// Wave (wrow, wcol): wrow = 16-row M block (0..3), wcol = 128-col half of the C output.
// Wh processed in chunks of 128: h = gelu(xn*W1+b1) -> LDS bf16 -> y += h*W2.
__global__ __launch_bounds__(256) void moeGemmK(const __bf16* __restrict__ xn,
                                                const __bf16* __restrict__ w1t,
                                                const float*  __restrict__ b1,
                                                const __bf16* __restrict__ w2t,
                                                const float*  __restrict__ b2,
                                                const int*    __restrict__ counts,
                                                const int*    __restrict__ lists,
                                                float* __restrict__ out) {
    __shared__ alignas(16) __bf16 aLds[TILE_M][CDIM + APAD];  // 33.8 KB
    __shared__ alignas(16) __bf16 hLds[TILE_M][WHC + APAD];   // 17.4 KB
    __shared__ int rowIdx[TILE_M];

    const int e   = blockIdx.x;
    const int cnt = counts[e];
    const int m0  = blockIdx.y * TILE_M;
    if (m0 >= cnt) return;

    const int tid  = threadIdx.x;
    const int lane = tid & 31;
    const int wid  = tid >> 5;
    const int wrow = wid >> 1;   // 0..3
    const int wcol = wid & 1;    // 0..1

    // gather this tile's (token,k) output rows
    if (tid < TILE_M) {
        const int r = m0 + tid;
        rowIdx[tid] = (r < cnt) ? lists[(size_t)e * NPAIR + r] : -1;
    }
    __syncthreads();

    // stage A tile [64 x 256] bf16 (gathered token rows) via CDNA5 async
    // global->LDS DMA (ASYNCcnt tracked, no VGPR staging).
    {
        const int r  = tid >> 2;           // 4 threads per row
        const int cq = (tid & 3) * 64;     // 64 halves each (128 B)
        const int orow = rowIdx[r];
        const int tok = (orow >= 0) ? (orow >> 1) : 0;   // KSEL == 2
        unsigned long long ga =
            (unsigned long long)(uintptr_t)(xn + (size_t)tok * CDIM + cq);
        unsigned la = (unsigned)(uintptr_t)&aLds[r][cq];
#pragma unroll
        for (int i = 0; i < 8; ++i) {
            asm volatile("global_load_async_to_lds_b128 %0, %1, off"
                         :: "v"(la + (unsigned)(i * 16)),
                            "v"(ga + (unsigned long long)(i * 16))
                         : "memory");
        }
        asm volatile("s_wait_asynccnt 0x0" ::: "memory");
    }
    __syncthreads();

    const __bf16* w1e = w1t + (size_t)e * WH * CDIM;
    const __bf16* w2e = w2t + (size_t)e * CDIM * WH;

    v8f acc[8] = {};   // y tile: rows wrow*16..+16, cols wcol*128..+128

#pragma unroll 1
    for (int ch = 0; ch < WH / WHC; ++ch) {
        const int whBase = ch * WHC;

        // ---- h chunk: [64 x 128] = xn[64x256] @ W1chunk ; this wave: 4 tiles
        v8f hacc[4] = {};
#pragma unroll
        for (int kt = 0; kt < CDIM / 32; ++kt) {
            const v16bf a = ldFragA(&aLds[0][0], CDIM + APAD, wrow * 16, kt * 32, lane);
#pragma unroll
            for (int ti = 0; ti < 4; ++ti) {
                const int n = whBase + (wcol * 4 + ti) * 16;
                const v16bf b = ldFragB(w1e, CDIM, n, kt * 32, lane);
                hacc[ti] = __builtin_amdgcn_wmma_f32_16x16x32_bf16(
                    false, a, false, b, (short)0, hacc[ti], false, false);
            }
        }
        // bias + exact GELU + store bf16 into hLds
#pragma unroll
        for (int ti = 0; ti < 4; ++ti) {
            const int ncol = (wcol * 4 + ti) * 16 + (lane & 15);
            const float bias = b1[e * WH + whBase + ncol];
#pragma unroll
            for (int r = 0; r < 8; ++r) {
                const int m = wrow * 16 + r + ((lane >> 4) << 3);
                float v = hacc[ti][r] + bias;
                v = 0.5f * v * (1.0f + erff(v * 0.70710678118654752f));
                hLds[m][ncol] = f2bf(v);
            }
        }
        __syncthreads();

        // ---- y accumulate: y[64x256] += h[64x128] @ W2chunk
#pragma unroll
        for (int k2 = 0; k2 < WHC / 32; ++k2) {
            const v16bf a = ldFragA(&hLds[0][0], WHC + APAD, wrow * 16, k2 * 32, lane);
#pragma unroll
            for (int ct = 0; ct < 8; ++ct) {
                const int n = wcol * 128 + ct * 16;
                const v16bf b = ldFragB(w2e, WH, n, whBase + k2 * 32, lane);
                acc[ct] = __builtin_amdgcn_wmma_f32_16x16x32_bf16(
                    false, a, false, b, (short)0, acc[ct], false, false);
            }
        }
        __syncthreads();   // hLds reused next chunk
    }

    // epilogue: + b2, scatter rows to out[t*K + k] (write-once -> non-temporal)
#pragma unroll
    for (int ct = 0; ct < 8; ++ct) {
        const int col = wcol * 128 + ct * 16 + (lane & 15);
        const float bias = b2[e * CDIM + col];
#pragma unroll
        for (int r = 0; r < 8; ++r) {
            const int m = wrow * 16 + r + ((lane >> 4) << 3);
            const int orow = rowIdx[m];
            if (orow >= 0) {
                __builtin_nontemporal_store(acc[ct][r] + bias,
                                            out + (size_t)orow * CDIM + col);
            }
        }
    }
}

// ---------------- launch ----------------

extern "C" void kernel_launch(void* const* d_in, const int* in_sizes, int n_in,
                              void* d_out, int out_size, void* d_ws, size_t ws_size,
                              hipStream_t stream) {
    const float* x    = (const float*)d_in[0];
    const float* rmsw = (const float*)d_in[1];
    const float* W1   = (const float*)d_in[2];
    const float* b1   = (const float*)d_in[3];
    const float* W2   = (const float*)d_in[4];
    const float* b2   = (const float*)d_in[5];
    const int*   eids = (const int*)d_in[6];
    float* out = (float*)d_out;

    // workspace layout (bytes):
    //   xn  : NTOK*CDIM*2          = 16,777,216
    //   w1t : NEXP*WH*CDIM*2       =  4,194,304
    //   w2t : NEXP*CDIM*WH*2       =  4,194,304
    //   cnt : 256 (8 ints, padded)
    //   list: NEXP*NPAIR*4         =  2,097,152     total ~27.3 MB
    char* ws = (char*)d_ws;
    size_t off = 0;
    __bf16* xn  = (__bf16*)(ws + off); off += (size_t)NTOK * CDIM * 2;
    __bf16* w1t = (__bf16*)(ws + off); off += (size_t)NEXP * WH * CDIM * 2;
    __bf16* w2t = (__bf16*)(ws + off); off += (size_t)NEXP * CDIM * WH * 2;
    int* counts = (int*)(ws + off);    off += 256;
    int* lists  = (int*)(ws + off);

    zeroCountsK<<<1, 32, 0, stream>>>(counts);
    rmsNormK<<<NTOK / 8, 256, 0, stream>>>(x, rmsw, xn);
    routeK<<<NPAIR / 256, 256, 0, stream>>>(eids, counts, lists);
    cvtW1K<<<(NEXP * WH * CDIM) / 256, 256, 0, stream>>>(W1, w1t);
    cvtW2K<<<(NEXP * CDIM * WH) / 256, 256, 0, stream>>>(W2, w2t);

    dim3 grid(NEXP, NPAIR / TILE_M);   // worst case: all pairs on one expert
    moeGemmK<<<grid, 256, 0, stream>>>(xn, w1t, b1, w2t, b2, counts, lists, out);
}
